// mam_66305705115939
// MI455X (gfx1250) — matrix operand order repaired and checked by reading, not compiled, not run on hardware
//
#include <hip/hip_runtime.h>
#include <hip/hip_bf16.h>

typedef __bf16 bf16_t;
typedef __attribute__((ext_vector_type(16))) __bf16 v16bf;
typedef __attribute__((ext_vector_type(8)))  __bf16 v8bf;
typedef __attribute__((ext_vector_type(8)))  float  v8f;

#define NIMG 8
#define CCH  64
#define HH   128
#define WW   128
#define HWPX 16384          // H*W
#define K9   576            // C*9 (GEMM K, also A row stride in global)
#define K9P  584            // padded LDS row stride: 1168B, /4 = 292 = 36 mod 64 banks
#define TP   72             // padded row stride for 64-wide T arrays (144B)
#define OU   32
#define EPSV 1e-5f

#define DEVFN static __device__ __forceinline__

DEVFN void splitbf(float x, bf16_t* h, bf16_t* l) {
  bf16_t hh = (bf16_t)x;
  *h = hh;
  *l = (bf16_t)(x - (float)hh);
}

DEVFN int iclamp(int v, int lo, int hi) { return v < lo ? lo : (v > hi ? hi : v); }

// A fragment (16x32 bf16, M x K): lane l holds row M = l%16.
// l<16 : K = kb+{0..7, 16..23};  l>=16 : K = kb+{8..15, 24..31}
DEVFN v16bf afrag(const bf16_t* Arow, int kb, int lane) {
  const bf16_t* p = Arow + kb + ((lane & 16) >> 1);   // +8 elems for upper half-wave
  v8bf c0 = *(const v8bf*)p;
  v8bf c1 = *(const v8bf*)(p + 16);
  v16bf a;
#pragma unroll
  for (int t = 0; t < 8; ++t) { a[t] = c0[t]; a[t + 8] = c1[t]; }
  return a;
}

// B fragment (32x16 bf16, K x N): lane l holds column N = l%16,
// l<16 : K = kb+{0..7, 16..23};  l>=16 : K = kb+{8..15, 24..31}, loaded as 2x b128
// (16B alignment only, so padded row strides are legal)
DEVFN v16bf bfrag(const bf16_t* Brow_n, int kb, int lane) {
  const bf16_t* p = Brow_n + kb + ((lane & 16) >> 1);
  v8bf c0 = *(const v8bf*)p;
  v8bf c1 = *(const v8bf*)(p + 16);
  v16bf b;
#pragma unroll
  for (int t = 0; t < 8; ++t) { b[t] = c0[t]; b[t + 8] = c1[t]; }
  return b;
}

// hi/lo 3-product bf16 emulation of an fp32 MAC tile
DEVFN v8f wmma3(v16bf ah, v16bf al, v16bf bh, v16bf bl, v8f c) {
  c = __builtin_amdgcn_wmma_f32_16x16x32_bf16(false, ah, false, bh, (short)0, c, false, false);
  c = __builtin_amdgcn_wmma_f32_16x16x32_bf16(false, ah, false, bl, (short)0, c, false, false);
  c = __builtin_amdgcn_wmma_f32_16x16x32_bf16(false, al, false, bh, (short)0, c, false, false);
  return c;
}

// ---------------- kernel 0: split all GEMM weights into bf16 hi/lo ----------------
__global__ void k_wprep(const float* dfw, const float* f2w, const float* cmw, const float* f3w,
                        bf16_t* dfh, bf16_t* dfl, bf16_t* f2h, bf16_t* f2l,
                        bf16_t* cmh, bf16_t* cml, bf16_t* f3h, bf16_t* f3l) {
  int i = blockIdx.x * blockDim.x + threadIdx.x;
  const int n_df = K9 * K9, n_f2 = OU * K9, n_cm = CCH * K9, n_f3 = OU * CCH;
  if (i < n_df) { splitbf(dfw[i], &dfh[i], &dfl[i]); return; }
  i -= n_df;
  if (i < n_f2) { splitbf(f2w[i], &f2h[i], &f2l[i]); return; }
  i -= n_f2;
  if (i < n_cm) { splitbf(cmw[i], &cmh[i], &cml[i]); return; }
  i -= n_cm;
  if (i < n_f3) { splitbf(f3w[i], &f3h[i], &f3l[i]); }
}

// ---------------- kernel 1: per-pixel alpha/beta (1x1 convs to 1 channel) --------
__global__ void k_alphabeta(const float* __restrict__ gr,
                            const float* __restrict__ aw, const float* __restrict__ ab,
                            const float* __restrict__ bw, const float* __restrict__ bb,
                            float* __restrict__ a_px, float* __restrict__ b_px) {
  int p = blockIdx.x * blockDim.x + threadIdx.x;
  if (p >= NIMG * HWPX) return;
  int n = p >> 14;
  int pix = p & (HWPX - 1);
  const float* g = gr + (size_t)n * CCH * HWPX + pix;
  float sa = 0.f, sb = 0.f;
#pragma unroll 8
  for (int c = 0; c < CCH; ++c) {
    float v = g[(size_t)c * HWPX];
    sa += v * aw[c];
    sb += v * bw[c];
  }
  a_px[p] = sa + ab[0];
  b_px[p] = sb + bb[0];
}

// ---------------- kernel 2: fus2 3x3 conv (implicit GEMM) -> cat channels 0..31 --
__global__ __launch_bounds__(128) void k_fus2(
    const float* __restrict__ gr, const float* __restrict__ gt,
    const float* __restrict__ a_px, const float* __restrict__ b_px,
    const bf16_t* __restrict__ f2h, const bf16_t* __restrict__ f2l,
    const float* __restrict__ f2b, const float* __restrict__ f2g, const float* __restrict__ f2bt,
    bf16_t* __restrict__ cath, bf16_t* __restrict__ catl) {
  __shared__ __align__(32) bf16_t Bh[32 * K9P];
  __shared__ __align__(32) bf16_t Bl[32 * K9P];
  const int tid = threadIdx.x;
  const int nimg = blockIdx.x >> 9;         // 512 tiles per image (HW/32)
  const int t = blockIdx.x & 511;
  const int y = t >> 2;                     // 4 tiles of 32 per row
  const int x0 = (t & 3) << 5;

  // Build B patches: x2 = gr + alpha*gt + beta, zero-padded 3x3, n-major LDS
  for (int idx = tid; idx < K9 * 32; idx += 128) {
    int k = idx >> 5, nn = idx & 31;
    int ic = k / 9, r = k - ic * 9;
    int dy = r / 3, dx = r - dy * 3;
    int gy = y + dy - 1, gx = x0 + nn + dx - 1;
    float v = 0.f;
    if ((unsigned)gy < (unsigned)HH && (unsigned)gx < (unsigned)WW) {
      size_t off = ((size_t)(nimg * CCH + ic)) * HWPX + gy * WW + gx;
      int pp = nimg * HWPX + gy * WW + gx;
      v = gr[off] + a_px[pp] * gt[off] + b_px[pp];
    }
    bf16_t h, l; splitbf(v, &h, &l);
    Bh[nn * K9P + k] = h;
    Bl[nn * K9P + k] = l;
  }
  __syncthreads();

  const int wave = tid >> 5, lane = tid & 31;
  const int mt = wave & 1, nt = wave >> 1;   // 2 M-tiles x 2 N-tiles
  const int mrow = mt * 16 + (lane & 15);
  const bf16_t* Ah = f2h + (size_t)mrow * K9;
  const bf16_t* Al = f2l + (size_t)mrow * K9;
  const bf16_t* Brh = Bh + (nt * 16 + (lane & 15)) * K9P;
  const bf16_t* Brl = Bl + (nt * 16 + (lane & 15)) * K9P;
  v8f acc = {};
  for (int kb = 0; kb < K9; kb += 32) {
    v16bf ah = afrag(Ah, kb, lane), al = afrag(Al, kb, lane);
    v16bf bh = bfrag(Brh, kb, lane), bl = bfrag(Brl, kb, lane);
    acc = wmma3(ah, al, bh, bl, acc);
  }
  const float rs = rsqrtf(1.f + EPSV);
  const int nn = nt * 16 + (lane & 15);
  const int gx = x0 + nn;
#pragma unroll
  for (int r2 = 0; r2 < 8; ++r2) {
    int m = mt * 16 + r2 + ((lane & 16) >> 1);
    float v = (acc[r2] + f2b[m]) * (f2g[m] * rs) + f2bt[m];
    v = fmaxf(v, 0.f);
    bf16_t h, l; splitbf(v, &h, &l);
    size_t off = ((size_t)(nimg * CCH + m)) * HWPX + y * WW + gx;
    cath[off] = h;
    catl[off] = l;
  }
}

// ---------------- kernel 3: fused df-conv (576ch) + dynamic 3x3 + fus3 1x1 -------
// 256 threads (8 waves), 32-px tile: halves weight re-streaming vs 16-px tiles.
__global__ __launch_bounds__(256) void k_dyn(
    const float* __restrict__ gr, const float* __restrict__ gt,
    const bf16_t* __restrict__ dfh, const bf16_t* __restrict__ dfl,
    const float* __restrict__ dfb,
    const bf16_t* __restrict__ f3h, const bf16_t* __restrict__ f3l,
    const float* __restrict__ f3b, const float* __restrict__ f3g, const float* __restrict__ f3bt,
    bf16_t* __restrict__ cath, bf16_t* __restrict__ catl) {
  __shared__ __align__(32) bf16_t Bh[32 * K9P];     // 74.75 KB (hi+lo)
  __shared__ __align__(32) bf16_t Bl[32 * K9P];
  __shared__ __align__(32) float  filt[K9 * 32];    // 73.7 KB, never hits HBM
  __shared__ __align__(32) bf16_t Th[32 * TP];
  __shared__ __align__(32) bf16_t Tl[32 * TP];
  const int tid = threadIdx.x;
  const int nimg = blockIdx.x >> 9;         // 512 tiles per image (HW/32)
  const int t = blockIdx.x & 511;
  const int y = t >> 2;
  const int x0 = (t & 3) << 5;

  // Phase 1: B patches of gr (zero pad)
  for (int idx = tid; idx < K9 * 32; idx += 256) {
    int k = idx >> 5, nn = idx & 31;
    int ic = k / 9, r = k - ic * 9;
    int dy = r / 3, dx = r - dy * 3;
    int gy = y + dy - 1, gx = x0 + nn + dx - 1;
    float v = 0.f;
    if ((unsigned)gy < (unsigned)HH && (unsigned)gx < (unsigned)WW)
      v = gr[((size_t)(nimg * CCH + ic)) * HWPX + gy * WW + gx];
    bf16_t h, l; splitbf(v, &h, &l);
    Bh[nn * K9P + k] = h;
    Bl[nn * K9P + k] = l;
  }
  __syncthreads();

  const int wave = tid >> 5, lane = tid & 31;
  const int nn = lane & 15;
  const bf16_t* B0h = Bh + nn * K9P;
  const bf16_t* B0l = Bl + nn * K9P;
  const bf16_t* B1h = Bh + (16 + nn) * K9P;
  const bf16_t* B1l = Bl + (16 + nn) * K9P;

  // Phase 2: GEMM 576x32x576 -> filt in LDS. A fragments reused across 2 N-tiles.
  for (int mt = wave; mt < 36; mt += 8) {
    const int mrow = mt * 16 + nn;
    const bf16_t* Ah = dfh + (size_t)mrow * K9;
    const bf16_t* Al = dfl + (size_t)mrow * K9;
    v8f acc0 = {}, acc1 = {};
    for (int kb = 0; kb < K9; kb += 32) {
      __builtin_prefetch(Ah + kb + 32, 0, 0);   // global_prefetch on streamed A rows
      v16bf ah = afrag(Ah, kb, lane), al = afrag(Al, kb, lane);
      v16bf b0h = bfrag(B0h, kb, lane), b0l = bfrag(B0l, kb, lane);
      acc0 = wmma3(ah, al, b0h, b0l, acc0);
      v16bf b1h = bfrag(B1h, kb, lane), b1l = bfrag(B1l, kb, lane);
      acc1 = wmma3(ah, al, b1h, b1l, acc1);
    }
#pragma unroll
    for (int r2 = 0; r2 < 8; ++r2) {
      int m = mt * 16 + r2 + ((lane & 16) >> 1);
      float bi = dfb[m];
      filt[m * 32 + nn] = acc0[r2] + bi;
      filt[m * 32 + 16 + nn] = acc1[r2] + bi;
    }
  }
  __syncthreads();

  // Phase 3: dynamic 3x3 conv on edge-padded gt, + gr, split for 1x1 GEMM
  for (int idx = tid; idx < CCH * 32; idx += 256) {
    int c = idx >> 5, px = idx & 31;
    int gx = x0 + px;
    const float* gtc = gt + ((size_t)(nimg * CCH + c)) * HWPX;
    float s = gr[((size_t)(nimg * CCH + c)) * HWPX + y * WW + gx];
#pragma unroll
    for (int dy = 0; dy < 3; ++dy) {
      int cy = iclamp(y + dy - 1, 0, HH - 1);
#pragma unroll
      for (int dx = 0; dx < 3; ++dx) {
        int cx = iclamp(gx + dx - 1, 0, WW - 1);
        s += gtc[cy * WW + cx] * filt[(c * 9 + dy * 3 + dx) * 32 + px];
      }
    }
    bf16_t h, l; splitbf(s, &h, &l);
    Th[px * TP + c] = h;
    Tl[px * TP + c] = l;
  }
  __syncthreads();

  // Phase 4: fus3 1x1 conv, M=32 K=64 N=32, waves 0..3
  if (wave < 4) {
    const int mt2 = wave & 1, nt2 = wave >> 1;
    const int mrow = mt2 * 16 + nn;
    const bf16_t* Ah = f3h + mrow * CCH;
    const bf16_t* Al = f3l + mrow * CCH;
    const bf16_t* B2h = Th + (nt2 * 16 + nn) * TP;
    const bf16_t* B2l = Tl + (nt2 * 16 + nn) * TP;
    v8f acc = {};
#pragma unroll
    for (int kb = 0; kb < CCH; kb += 32) {
      v16bf ah = afrag(Ah, kb, lane), al = afrag(Al, kb, lane);
      v16bf bh = bfrag(B2h, kb, lane), bl = bfrag(B2l, kb, lane);
      acc = wmma3(ah, al, bh, bl, acc);
    }
    const float rs = rsqrtf(1.f + EPSV);
    const int gx = x0 + nt2 * 16 + nn;
#pragma unroll
    for (int r2 = 0; r2 < 8; ++r2) {
      int m = mt2 * 16 + r2 + ((lane & 16) >> 1);
      float v = (acc[r2] + f3b[m]) * (f3g[m] * rs) + f3bt[m];
      v = fmaxf(v, 0.f);
      bf16_t h, l; splitbf(v, &h, &l);
      size_t off = ((size_t)(nimg * CCH + OU + m)) * HWPX + y * WW + gx;
      cath[off] = h;
      catl[off] = l;
    }
  }
}

// ---------------- kernel 4: comb 3x3 conv over cat (implicit GEMM) -> out --------
__global__ __launch_bounds__(128) void k_comb(
    const bf16_t* __restrict__ cath, const bf16_t* __restrict__ catl,
    const bf16_t* __restrict__ cwh, const bf16_t* __restrict__ cwl,
    const float* __restrict__ cb, const float* __restrict__ cg, const float* __restrict__ cbt,
    float* __restrict__ out) {
  __shared__ __align__(32) bf16_t Bh[32 * K9P];
  __shared__ __align__(32) bf16_t Bl[32 * K9P];
  const int tid = threadIdx.x;
  const int nimg = blockIdx.x >> 9;
  const int t = blockIdx.x & 511;
  const int y = t >> 2;
  const int x0 = (t & 3) << 5;

  for (int idx = tid; idx < K9 * 32; idx += 128) {
    int k = idx >> 5, nn = idx & 31;
    int ic = k / 9, r = k - ic * 9;
    int dy = r / 3, dx = r - dy * 3;
    int gy = y + dy - 1, gx = x0 + nn + dx - 1;
    bf16_t h = (bf16_t)0.f, l = (bf16_t)0.f;
    if ((unsigned)gy < (unsigned)HH && (unsigned)gx < (unsigned)WW) {
      size_t off = ((size_t)(nimg * CCH + ic)) * HWPX + gy * WW + gx;
      h = cath[off];
      l = catl[off];
    }
    Bh[nn * K9P + k] = h;
    Bl[nn * K9P + k] = l;
  }
  __syncthreads();

  const int wave = tid >> 5, lane = tid & 31;   // wave = M-tile (4 of them)
  const int mrow = wave * 16 + (lane & 15);
  const bf16_t* Ah = cwh + (size_t)mrow * K9;
  const bf16_t* Al = cwl + (size_t)mrow * K9;
  const bf16_t* B0h = Bh + (lane & 15) * K9P;
  const bf16_t* B0l = Bl + (lane & 15) * K9P;
  const bf16_t* B1h = Bh + (16 + (lane & 15)) * K9P;
  const bf16_t* B1l = Bl + (16 + (lane & 15)) * K9P;
  v8f acc0 = {}, acc1 = {};
  for (int kb = 0; kb < K9; kb += 32) {
    v16bf ah = afrag(Ah, kb, lane), al = afrag(Al, kb, lane);
    v16bf b0h = bfrag(B0h, kb, lane), b0l = bfrag(B0l, kb, lane);
    acc0 = wmma3(ah, al, b0h, b0l, acc0);
    v16bf b1h = bfrag(B1h, kb, lane), b1l = bfrag(B1l, kb, lane);
    acc1 = wmma3(ah, al, b1h, b1l, acc1);
  }
  const float rs = rsqrtf(1.f + EPSV);
#pragma unroll
  for (int r2 = 0; r2 < 8; ++r2) {
    int m = wave * 16 + r2 + ((lane & 16) >> 1);
    float sc = cg[m] * rs, bo = cbt[m], bi = cb[m];
    float v0 = fmaxf((acc0[r2] + bi) * sc + bo, 0.f);
    float v1 = fmaxf((acc1[r2] + bi) * sc + bo, 0.f);
    size_t base = ((size_t)(nimg * CCH + m)) * HWPX + y * WW + x0;
    out[base + (lane & 15)] = v0;
    out[base + 16 + (lane & 15)] = v1;
  }
}

extern "C" void kernel_launch(void* const* d_in, const int* in_sizes, int n_in,
                              void* d_out, int out_size, void* d_ws, size_t ws_size,
                              hipStream_t stream) {
  (void)in_sizes; (void)n_in; (void)out_size; (void)ws_size;
  const float* gr   = (const float*)d_in[0];
  const float* gt   = (const float*)d_in[1];
  const float* aw   = (const float*)d_in[2];
  const float* ab   = (const float*)d_in[3];
  const float* bw   = (const float*)d_in[4];
  const float* bb   = (const float*)d_in[5];
  const float* f2w  = (const float*)d_in[6];
  const float* f2b  = (const float*)d_in[7];
  const float* f2g  = (const float*)d_in[8];
  const float* f2bt = (const float*)d_in[9];
  const float* dfw  = (const float*)d_in[10];
  const float* dfb  = (const float*)d_in[11];
  const float* f3w  = (const float*)d_in[12];
  const float* f3b  = (const float*)d_in[13];
  const float* f3g  = (const float*)d_in[14];
  const float* f3bt = (const float*)d_in[15];
  const float* cw   = (const float*)d_in[16];
  const float* cbias= (const float*)d_in[17];
  const float* cg   = (const float*)d_in[18];
  const float* cbt  = (const float*)d_in[19];
  float* out = (float*)d_out;

  char* w = (char*)d_ws;
  size_t off = 0;
  auto alloc = [&](size_t bytes) -> char* {
    char* p = w + off;
    off += (bytes + 255) & ~(size_t)255;
    return p;
  };
  const size_t catElems = (size_t)NIMG * CCH * HWPX;
  bf16_t* cath = (bf16_t*)alloc(catElems * sizeof(bf16_t));
  bf16_t* catl = (bf16_t*)alloc(catElems * sizeof(bf16_t));
  float*  a_px = (float*)alloc((size_t)NIMG * HWPX * sizeof(float));
  float*  b_px = (float*)alloc((size_t)NIMG * HWPX * sizeof(float));
  bf16_t* dfh = (bf16_t*)alloc((size_t)K9 * K9 * sizeof(bf16_t));
  bf16_t* dfl = (bf16_t*)alloc((size_t)K9 * K9 * sizeof(bf16_t));
  bf16_t* f2h = (bf16_t*)alloc((size_t)OU * K9 * sizeof(bf16_t));
  bf16_t* f2l = (bf16_t*)alloc((size_t)OU * K9 * sizeof(bf16_t));
  bf16_t* cmh = (bf16_t*)alloc((size_t)CCH * K9 * sizeof(bf16_t));
  bf16_t* cml = (bf16_t*)alloc((size_t)CCH * K9 * sizeof(bf16_t));
  bf16_t* f3h = (bf16_t*)alloc((size_t)OU * CCH * sizeof(bf16_t));
  bf16_t* f3l = (bf16_t*)alloc((size_t)OU * CCH * sizeof(bf16_t));

  const int nWeights = K9 * K9 + OU * K9 + CCH * K9 + OU * CCH;
  k_wprep<<<(nWeights + 255) / 256, 256, 0, stream>>>(
      dfw, f2w, cw, f3w, dfh, dfl, f2h, f2l, cmh, cml, f3h, f3l);

  k_alphabeta<<<(NIMG * HWPX) / 256, 256, 0, stream>>>(gr, aw, ab, bw, bb, a_px, b_px);

  k_fus2<<<NIMG * (HWPX / 32), 128, 0, stream>>>(
      gr, gt, a_px, b_px, f2h, f2l, f2b, f2g, f2bt, cath, catl);

  k_dyn<<<NIMG * (HWPX / 32), 256, 0, stream>>>(
      gr, gt, dfh, dfl, dfb, f3h, f3l, f3b, f3g, f3bt, cath, catl);

  k_comb<<<NIMG * (HWPX / 32), 128, 0, stream>>>(
      cath, catl, cmh, cml, cbias, cg, cbt, out);
}
